// GraphQNN__gen246_65481071397259
// MI455X (gfx1250) — matrix operand order repaired and checked by reading, not compiled, not run on hardware
//
#include <hip/hip_runtime.h>
#include <hip/hip_bf16.h>

typedef __attribute__((ext_vector_type(16))) __bf16 v16bf;
typedef __attribute__((ext_vector_type(8)))  __bf16 v8bf;
typedef __attribute__((ext_vector_type(8)))  float  v8f;
typedef __attribute__((ext_vector_type(4)))  float  v4f;

namespace gq {
constexpr int N  = 16384;   // nodes
constexpr int D  = 128;     // feature dim
constexpr int C  = 16;      // classes
constexpr int BM = 64;      // output rows per workgroup
constexpr int BK = 32;      // K-step (one bf16 WMMA depth)
constexpr int KT = N / BK;  // 512 K-steps
}

// ---------------------------------------------------------------------------
// Pack H (N x 128 fp32, row-major) into bf16 fragment order:
//   dst[kt*256 + nt*32 + lane] = v16bf with elem e = H[kt*32 + 16*(lane>>4)+e,
//                                                      nt*16 + (lane&15)]
// This is the contiguous-k B-operand order this GEMM uses (A in LDS matches), so the GEMM can load each
// 32x16 B fragment as one contiguous 32B global (L2) load. 4MB, L2-resident.
// ---------------------------------------------------------------------------
__global__ __launch_bounds__(256) void pack_frag(const float* __restrict__ src,
                                                 v16bf* __restrict__ dst) {
  using namespace gq;
  const int id   = blockIdx.x * blockDim.x + threadIdx.x;  // 0 .. KT*256-1
  const int lane = id & 31;
  const int nt   = (id >> 5) & 7;
  const int kt   = id >> 8;
  const int n    = nt * 16 + (lane & 15);
  const int k0   = kt * 32 + 16 * (lane >> 4);
  v16bf p;
#pragma unroll
  for (int e = 0; e < 16; ++e) p[e] = (__bf16)src[(size_t)(k0 + e) * D + n];
  dst[id] = p;
}

// ---------------------------------------------------------------------------
// agg = A(16384x16384 fp32, streamed NT from HBM) @ H (pre-packed bf16 frags,
// L2-resident). 8 waves/WG in a 4x2 grid; each wave owns a 16x64 strip
// (4 v8f accumulators). A is staged through double-buffered LDS in fragment
// order (shared by the two column-half waves); B fragments are loaded directly
// from global, software-pipelined one K-step ahead.
// ---------------------------------------------------------------------------
__global__ __launch_bounds__(256) void agg_gemm(const float* __restrict__ A,
                                                const v16bf* __restrict__ Bf,
                                                float* __restrict__ out) {
  using namespace gq;
  __shared__ __align__(32) __bf16 sA[2][BM * BK];  // 4KB per buffer

  const int t    = threadIdx.x;
  const int lane = t & 31;
  const int w    = t >> 5;   // wave 0..7
  const int wm   = w >> 1;   // 0..3 : 16-row subtile
  const int wn   = w & 1;    // 0..1 : 64-col half
  const int r0   = blockIdx.x * BM;

  // A staging geometry: thread t loads row aRow, cols aC0..aC0+7 (fp32),
  // writes one 16B bf16 chunk at the fragment position.
  const int aRow = t >> 2;             // 0..63
  const int aC0  = (t & 3) * 8;        // 0,8,16,24
  const int aH   = aC0 >> 4;
  const int aE0  = (aC0 & 8) ? 8 : 0;
  const int aDst = (aRow >> 4) * 512 + (((aH << 4) + (aRow & 15)) << 4) + aE0;
  const size_t aBase = (size_t)(r0 + aRow) * N + aC0;

  v4f aReg0, aReg1;

  auto gloadA = [&](int kt) {
    const float* ap = A + aBase + (size_t)kt * BK;
    aReg0 = __builtin_nontemporal_load((const v4f*)ap);      // stream adjacency,
    aReg1 = __builtin_nontemporal_load((const v4f*)ap + 1);  // keep L2 for H/W
  };
  auto stageA = [&](int buf) {
    v8bf pa;
#pragma unroll
    for (int j = 0; j < 4; ++j) { pa[j] = (__bf16)aReg0[j]; pa[4 + j] = (__bf16)aReg1[j]; }
    *(v8bf*)&sA[buf][aDst] = pa;
  };
  auto gloadB = [&](int kt, v16bf* b) {
#pragma unroll
    for (int j = 0; j < 4; ++j)
      b[j] = Bf[(size_t)kt * 256 + (wn * 4 + j) * 32 + lane];
  };

  v8f acc[4];
#pragma unroll
  for (int i = 0; i < 4; ++i)
#pragma unroll
    for (int e = 0; e < 8; ++e) acc[i][e] = 0.0f;

  v16bf bcur[4], bnxt[4];
  gloadA(0);
  gloadB(0, bcur);
  stageA(0);
  __syncthreads();

#pragma unroll 2
  for (int kt = 0; kt < KT; ++kt) {
    const int cur = kt & 1;
    if (kt + 1 < KT) {
      gloadA(kt + 1);        // next adjacency tile (HBM, NT)
      gloadB(kt + 1, bnxt);  // next B fragments (L2)
    }

    const v16bf afrag = *(const v16bf*)&sA[cur][wm * 512 + lane * 16];
#pragma unroll
    for (int j = 0; j < 4; ++j)
      acc[j] = __builtin_amdgcn_wmma_f32_16x16x32_bf16(
          false, afrag, false, bcur[j], (short)0, acc[j], false, false);

    if (kt + 1 < KT) {
      stageA(1 - cur);
#pragma unroll
      for (int j = 0; j < 4; ++j) bcur[j] = bnxt[j];
    }
    __syncthreads();
  }

  // C/D layout: VGPR j -> M = j + 8*(lane>>4); N = lane&15
  const int hlf  = lane >> 4;
  const int nloc = lane & 15;
#pragma unroll
  for (int jt = 0; jt < 4; ++jt) {
    const int col = wn * 64 + jt * 16 + nloc;
#pragma unroll
    for (int r = 0; r < 8; ++r) {
      const int row = r0 + wm * 16 + hlf * 8 + r;
      out[row * D + col] = acc[jt][r];
    }
  }
}

// ---------------------------------------------------------------------------
// h[n,o] = relu( sum_i agg[n,i] * W[type[n], o, i] )   (exact fp32)
// ---------------------------------------------------------------------------
__global__ __launch_bounds__(128) void type_transform(const float* __restrict__ agg,
                                                      const int* __restrict__ types,
                                                      const float* __restrict__ W,
                                                      float* __restrict__ h) {
  using namespace gq;
  __shared__ __align__(16) float row[D];
  const int n = blockIdx.x;
  const int o = threadIdx.x;
  row[o] = agg[(size_t)n * D + o];
  __syncthreads();

  const int tp = types[n];
  const v4f* w4 = (const v4f*)(W + ((size_t)tp * D + o) * D);
  const v4f* r4 = (const v4f*)row;
  float acc = 0.0f;
#pragma unroll
  for (int i = 0; i < D / 4; ++i) {
    const v4f a = r4[i], b = w4[i];
    acc += a[0] * b[0] + a[1] * b[1] + a[2] * b[2] + a[3] * b[3];
  }
  h[(size_t)n * D + o] = fmaxf(acc, 0.0f);
}

// ---------------------------------------------------------------------------
// logits[n,c] = h[n,:] . Wc[c,:] + bc[c]   (exact fp32)
// ---------------------------------------------------------------------------
__global__ __launch_bounds__(256) void classifier(const float* __restrict__ h,
                                                  const float* __restrict__ Wc,
                                                  const float* __restrict__ bc,
                                                  float* __restrict__ out) {
  using namespace gq;
  const int idx = blockIdx.x * blockDim.x + threadIdx.x;  // n*C + c
  const int n = idx >> 4;
  const int c = idx & (C - 1);
  const v4f* h4 = (const v4f*)(h + (size_t)n * D);
  const v4f* w4 = (const v4f*)(Wc + (size_t)c * D);
  float acc = bc[c];
#pragma unroll
  for (int i = 0; i < D / 4; ++i) {
    const v4f a = h4[i], b = w4[i];
    acc += a[0] * b[0] + a[1] * b[1] + a[2] * b[2] + a[3] * b[3];
  }
  out[idx] = acc;
}

extern "C" void kernel_launch(void* const* d_in, const int* in_sizes, int n_in,
                              void* d_out, int out_size, void* d_ws, size_t ws_size,
                              hipStream_t stream) {
  using namespace gq;
  const float* x   = (const float*)d_in[0];
  const int*   nt  = (const int*)d_in[1];
  const float* adj = (const float*)d_in[2];
  const float* W1  = (const float*)d_in[3];
  const float* W2  = (const float*)d_in[4];
  const float* Wc  = (const float*)d_in[5];
  const float* bcv = (const float*)d_in[6];
  float* out = (float*)d_out;

  float* agg = (float*)d_ws;                 // 8MB  fp32 aggregation buffer
  float* h   = agg + (size_t)N * D;          // 8MB  fp32 hidden
  v16bf* Hf  = (v16bf*)(h + (size_t)N * D);  // 4MB  bf16 fragment-packed H

  // layer 1
  pack_frag<<<(N * D / 16) / 256, 256, 0, stream>>>(x, Hf);
  agg_gemm<<<N / BM, 256, 0, stream>>>(adj, Hf, agg);
  type_transform<<<N, D, 0, stream>>>(agg, nt, W1, h);
  // layer 2
  pack_frag<<<(N * D / 16) / 256, 256, 0, stream>>>(h, Hf);
  agg_gemm<<<N / BM, 256, 0, stream>>>(adj, Hf, agg);
  type_transform<<<N, D, 0, stream>>>(agg, nt, W2, h);
  // classifier
  classifier<<<(N * C) / 256, 256, 0, stream>>>(h, Wc, bcv, out);
}
